// DiffusionPriorNetwork_72069551227054
// MI455X (gfx1250) — matrix operand order repaired and checked by reading, not compiled, not run on hardware
//
#include <hip/hip_runtime.h>
#include <hip/hip_bf16.h>
#include <math.h>

// ---------------- constants ----------------
#define BB   32
#define DD   1024
#define LL   256
#define TEE  768
#define HH   16
#define DHH  64
#define SEQ  260          // L + 4 tokens
#define NJ   261          // SEQ + 1 (null kv)
#define NTOK (BB*SEQ)     // 8320
#define FFN  4096
#define ROTD 32
#define NEGF (-3.4028235e38f)

// ---------------- WMMA types ----------------
typedef __attribute__((ext_vector_type(16))) __bf16 v16bf;
typedef __attribute__((ext_vector_type(8)))  float  v8f;

union BfBits { unsigned short u; __bf16 b; };

__device__ __forceinline__ unsigned short f2bf_bits(float f) {
    union { float f; unsigned int u; } x; x.f = f;
    unsigned int r = x.u + 0x7FFFu + ((x.u >> 16) & 1u);   // RNE
    return (unsigned short)(r >> 16);
}

__device__ __forceinline__ float reduceSum(float v, float* red, int nthr) {
    int t = threadIdx.x;
    red[t] = v; __syncthreads();
    for (int s = nthr >> 1; s > 0; s >>= 1) {
        if (t < s) red[t] += red[t + s];
        __syncthreads();
    }
    float r = red[0]; __syncthreads();
    return r;
}
__device__ __forceinline__ float reduceMax(float v, float* red, int nthr) {
    int t = threadIdx.x;
    red[t] = v; __syncthreads();
    for (int s = nthr >> 1; s > 0; s >>= 1) {
        if (t < s) red[t] = fmaxf(red[t], red[t + s]);
        __syncthreads();
    }
    float r = red[0]; __syncthreads();
    return r;
}

__device__ __forceinline__ float siluf(float x) { return x / (1.f + expf(-x)); }

// ---------------- generic WMMA GEMM ----------------
// C[M,N] = act(alpha * A[M,K] @ B[K,N] + bias) (+= C if accum)
// A row-major lda, B row-major ldb, C row-major ldc.
// K multiple of 32. N multiple of 128 for every launch in this pipeline.
#define BM 64
#define BN 128
#define KS 32

__global__ __launch_bounds__(256)
void gemm_wmma_bf16(const float* __restrict__ A, const float* __restrict__ Bm,
                    const float* __restrict__ bias, float* __restrict__ Cm,
                    int M, int N, int K, int lda, int ldb, int ldc,
                    float alpha, int act, int accum)
{
    // async-staged raw f32 tiles
    __shared__ __align__(16) float sAf[BM * KS];            // 8 KB  [m][k]
    __shared__ __align__(16) float sBf[KS * BN];            // 16 KB [k][n]
    // converted bf16 tiles consumed by WMMA
    __shared__ __align__(16) unsigned short sA[BM * KS];    // 4 KB  [m][k]
    __shared__ __align__(16) unsigned short sB[BN * KS];    // 8 KB  [n][k] (transposed)

    const int tid  = threadIdx.x;
    const int wave = tid >> 5;
    const int lane = tid & 31;
    const int wm   = wave >> 2;     // 0..1
    const int wn   = wave & 3;      // 0..3
    const int m0   = blockIdx.y * BM;
    const int n0   = blockIdx.x * BN;

    v8f z = {};
    v8f acc[2][2];
    acc[0][0] = z; acc[0][1] = z; acc[1][0] = z; acc[1][1] = z;

    const int r  = lane & 15;
    const int ka = (lane >> 4) * 16;   // A fragment K offset (contiguous 16)
    const int kb = (lane >> 4) * 16;   // B fragment K offset

    const bool fullM = (m0 + BM <= M);  // uniform over the block

    // per-lane async staging coordinates
    const int arow = tid >> 4;            // 0..15? no: 256/16 -> 0..15 per 16 lanes... (tid>>4) in 0..15 covers 16 rows/iter? see below
    // A: each thread moves 2 f32 (b64):  row = tid>>4 (0..15)*? -> 256thr x 8B = 512 of the 2048 f32 (rows 0..15)
    //    element index = tid*2: row = (tid*2)>>5 = tid>>4, kk = (tid*2)&31
    const int a_k  = (tid * 2) & 31;
    const int a_m  = (tid * 2) >> 5;
    // B: each thread moves 4 f32 (b128): element index = tid*4: kk = tid>>5, nn = (tid*4)&127
    const int b_k  = tid >> 5;
    const int b_n  = (tid * 4) & 127;

    const unsigned ldsA = (unsigned)(unsigned long long)(&sAf[tid * 2]);
    const unsigned ldsB = (unsigned)(unsigned long long)(&sBf[tid * 4]);

    for (int k0 = 0; k0 < K; k0 += KS) {
        if (fullM) {
            // ---- CDNA5 async global -> LDS staging (raw f32) ----
            unsigned long long aaddr =
                (unsigned long long)(A  + (long)(m0 + a_m) * lda + (k0 + a_k));
            unsigned long long baddr =
                (unsigned long long)(Bm + (long)(k0 + b_k) * ldb + (n0 + b_n));
            asm volatile("global_load_async_to_lds_b64 %0, %1, off"
                         :: "v"(ldsA), "v"(aaddr) : "memory");
            asm volatile("global_load_async_to_lds_b128 %0, %1, off"
                         :: "v"(ldsB), "v"(baddr) : "memory");
            asm volatile("s_wait_asynccnt 0x0" ::: "memory");
            __syncthreads();

            // ---- LDS f32 -> LDS bf16 convert (A: same layout, packed pairs) ----
            #pragma unroll
            for (int e = 0; e < 2; ++e) {
                int off = e * 1024 + tid * 4;                 // multiple of 4
                float4 fv = *(const float4*)&sAf[off];
                unsigned lo = (unsigned)f2bf_bits(fv.x) | ((unsigned)f2bf_bits(fv.y) << 16);
                unsigned hi = (unsigned)f2bf_bits(fv.z) | ((unsigned)f2bf_bits(fv.w) << 16);
                *(uint2*)&sA[off] = make_uint2(lo, hi);
            }
            // ---- B convert + transpose: pack k-pairs along n groups of 4 ----
            #pragma unroll
            for (int e = 0; e < 2; ++e) {
                int u  = e * 256 + tid;       // 0..511 units
                int kp = u >> 5;              // 0..15  (pair of k rows: 2kp, 2kp+1)
                int ng = u & 31;              // 0..31  (group of 4 n)
                const float4 r0 = *(const float4*)&sBf[(2*kp    ) * BN + ng * 4];
                const float4 r1 = *(const float4*)&sBf[(2*kp + 1) * BN + ng * 4];
                unsigned p0 = (unsigned)f2bf_bits(r0.x) | ((unsigned)f2bf_bits(r1.x) << 16);
                unsigned p1 = (unsigned)f2bf_bits(r0.y) | ((unsigned)f2bf_bits(r1.y) << 16);
                unsigned p2 = (unsigned)f2bf_bits(r0.z) | ((unsigned)f2bf_bits(r1.z) << 16);
                unsigned p3 = (unsigned)f2bf_bits(r0.w) | ((unsigned)f2bf_bits(r1.w) << 16);
                *(unsigned*)&sB[(ng*4 + 0) * KS + 2*kp] = p0;
                *(unsigned*)&sB[(ng*4 + 1) * KS + 2*kp] = p1;
                *(unsigned*)&sB[(ng*4 + 2) * KS + 2*kp] = p2;
                *(unsigned*)&sB[(ng*4 + 3) * KS + 2*kp] = p3;
            }
        } else {
            // ---- guarded scalar path (M=32 launches and the last row block of M=2080 launches hit this) ----
            #pragma unroll
            for (int e = 0; e < (BM*KS)/256; ++e) {
                int idx  = e * 256 + tid;
                int mrow = idx >> 5;
                int kk   = idx & 31;
                int gm   = m0 + mrow;
                float v  = (gm < M) ? A[(long)gm * lda + (k0 + kk)] : 0.f;
                sA[mrow * KS + kk] = f2bf_bits(v);
            }
            #pragma unroll
            for (int e = 0; e < (BN*KS)/256; ++e) {
                int idx = e * 256 + tid;
                int kk  = idx >> 7;
                int nn  = idx & 127;
                int gn  = n0 + nn;
                float v = (gn < N) ? Bm[(long)(k0 + kk) * ldb + gn] : 0.f;
                sB[nn * KS + kk] = f2bf_bits(v);
            }
        }
        __syncthreads();

        // ---- WMMA on bf16 tiles ----
        #pragma unroll
        for (int ti = 0; ti < 2; ++ti) {
            v16bf afrag;
            const unsigned short* pa = &sA[(wm*32 + ti*16 + r) * KS + ka];
            #pragma unroll
            for (int e2 = 0; e2 < 8; ++e2) {
                BfBits c0; c0.u = pa[e2];      afrag[e2]     = c0.b;
                BfBits c1; c1.u = pa[e2 + 8]; afrag[e2 + 8] = c1.b;
            }
            #pragma unroll
            for (int tj = 0; tj < 2; ++tj) {
                v16bf bfrag;
                const unsigned short* pb = &sB[(wn*32 + tj*16 + r) * KS + kb];
                #pragma unroll
                for (int e2 = 0; e2 < 16; ++e2) {
                    BfBits c2; c2.u = pb[e2]; bfrag[e2] = c2.b;
                }
                acc[ti][tj] = __builtin_amdgcn_wmma_f32_16x16x32_bf16(
                    false, afrag, false, bfrag, (short)0, acc[ti][tj], false, false);
            }
        }
        __syncthreads();
    }

    const int mlane = (lane >> 4) * 8;
    const int nlane = lane & 15;
    #pragma unroll
    for (int ti = 0; ti < 2; ++ti)
    #pragma unroll
    for (int tj = 0; tj < 2; ++tj)
    #pragma unroll
    for (int v = 0; v < 8; ++v) {
        int gm = m0 + wm*32 + ti*16 + mlane + v;
        int gn = n0 + wn*32 + tj*16 + nlane;
        if (gm < M && gn < N) {
            float rr = acc[ti][tj][v] * alpha;
            if (bias) rr += bias[gn];
            if (act == 1)      rr = fmaxf(rr, 0.f);
            else if (act == 2) rr = siluf(rr);
            long off = (long)gm * ldc + gn;
            if (accum) rr += Cm[off];
            Cm[off] = rr;
        }
    }
}

// ---------------- LayerNorm (D=1024, 256 thr, 4 elem/thr) ----------------
__global__ __launch_bounds__(256)
void ln_kernel(const float* __restrict__ in, const float* __restrict__ g,
               const float* __restrict__ res, float* __restrict__ out,
               long in_stride, long res_stride, long out_stride, int stable)
{
    __shared__ float red[256];
    int row = blockIdx.x, t = threadIdx.x;
    const float* x = in + (long)row * in_stride;
    float v[4];
    #pragma unroll
    for (int e = 0; e < 4; ++e) v[e] = x[t + e*256];
    if (stable) {
        float lm = fmaxf(fmaxf(v[0], v[1]), fmaxf(v[2], v[3]));
        float mx = reduceMax(lm, red, 256);
        #pragma unroll
        for (int e = 0; e < 4; ++e) v[e] /= mx;
    }
    float s = v[0] + v[1] + v[2] + v[3];
    float mean = reduceSum(s, red, 256) * (1.f / 1024.f);
    float sq = 0.f;
    #pragma unroll
    for (int e = 0; e < 4; ++e) { float d = v[e] - mean; sq += d * d; }
    float var = reduceSum(sq, red, 256) * (1.f / 1024.f);
    float inv = rsqrtf(var + 1e-5f);
    #pragma unroll
    for (int e = 0; e < 4; ++e) {
        int c = t + e*256;
        float o = (v[e] - mean) * inv * g[c];
        if (res) o += res[(long)row * res_stride + c];
        out[(long)row * out_stride + c] = o;
    }
}

// ---------------- sinusoidal time embedding ----------------
__global__ void se_kernel(const float* __restrict__ ts, float* __restrict__ se)
{
    int b = blockIdx.x, t = threadIdx.x;
    float tv = ts[b];
    for (int e = 0; e < 2; ++e) {
        int j = t + e*256;                       // 0..511
        float f = expf((float)j * (-logf(10000.f) / 511.f));
        float a = tv * f;
        se[(long)b*DD + j]       = sinf(a);
        se[(long)b*DD + 512 + j] = cosf(a);
    }
}

// ---------------- token assembly ----------------
__global__ __launch_bounds__(256)
void assemble_kernel(const float* __restrict__ te, const float* __restrict__ text_embed,
                     const float* __restrict__ temb, const float* __restrict__ image_embed,
                     const float* __restrict__ lq, float* __restrict__ X)
{
    int i = blockIdx.x, b = blockIdx.y, t = threadIdx.x;
    #pragma unroll
    for (int e = 0; e < 4; ++e) {
        int d = t + e*256;
        float v;
        if (i < LL)        v = te[((long)b*LL + i)*DD + d];
        else if (i == LL)  v = text_embed[(long)b*DD + d];
        else if (i == LL+1)v = temb[(long)b*DD + d];
        else if (i == LL+2)v = image_embed[(long)b*DD + d];
        else               v = lq[d];
        X[((long)b*SEQ + i)*DD + d] = v;
    }
}

// ---------------- rotary + l2norm on Q (in place) ----------------
__global__ void rotary_q_kernel(float* __restrict__ Q)
{
    __shared__ float red[64];
    int h = blockIdx.x, i = blockIdx.y, b = blockIdx.z, t = threadIdx.x;
    float* row = Q + (((long)b*SEQ + i)*HH + h)*DHH;
    float v = row[t];
    if (t < ROTD) {
        int p = t >> 1;
        float invf = expf(-(float)(2*p) / ROTD * logf(10000.f));
        float ang = (float)i * invf;
        float c = cosf(ang), s = sinf(ang);
        float other = __shfl_xor(v, 1, 32);
        v = (t & 1) ? (v*c + other*s) : (v*c - other*s);
    }
    float nsq = reduceSum(v*v, red, 64);
    v = v / fmaxf(sqrtf(nsq), 1e-12f) * 4.0f;
    row[t] = v;
}

// ---------------- split KV, rotary + l2norm K, place at j=i+1 ----------------
__global__ void rotary_kv_kernel(const float* __restrict__ KV,
                                 float* __restrict__ Kb, float* __restrict__ Vb)
{
    __shared__ float red[64];
    int i = blockIdx.x, b = blockIdx.y, t = threadIdx.x;
    const float* kvrow = KV + ((long)b*SEQ + i) * (2*DHH);
    float kv_ = kvrow[t];
    float vv  = kvrow[DHH + t];
    if (t < ROTD) {
        int p = t >> 1;
        float invf = expf(-(float)(2*p) / ROTD * logf(10000.f));
        float ang = (float)i * invf;
        float c = cosf(ang), s = sinf(ang);
        float other = __shfl_xor(kv_, 1, 32);
        kv_ = (t & 1) ? (kv_*c + other*s) : (kv_*c - other*s);
    }
    float nsq = reduceSum(kv_*kv_, red, 64);
    kv_ = kv_ / fmaxf(sqrtf(nsq), 1e-12f) * 4.0f;
    long off = ((long)b*NJ + (i + 1)) * DHH + t;
    Kb[off] = kv_;
    Vb[off] = vv;
}

// ---------------- null kv at j=0 (K l2-normed*4, V raw) ----------------
__global__ void nullkv_kernel(const float* __restrict__ nkv,
                              float* __restrict__ Kb, float* __restrict__ Vb)
{
    __shared__ float red[64];
    int b = blockIdx.x, t = threadIdx.x;
    float nk = nkv[t];
    float nv = nkv[DHH + t];
    float nsq = reduceSum(nk*nk, red, 64);
    nk = nk / fmaxf(sqrtf(nsq), 1e-12f) * 4.0f;
    Kb[(long)b*NJ*DHH + t] = nk;
    Vb[(long)b*NJ*DHH + t] = nv;
}

// ---------------- fused attention row: scores + relpos bias + mask + softmax + V ----------------
__global__ void attn_kernel(const float* __restrict__ Q, const float* __restrict__ Kb,
                            const float* __restrict__ Vb, const float* __restrict__ relpos,
                            float* __restrict__ O)
{
    __shared__ float qs[DHH];
    __shared__ float sc[NJ + 3];
    __shared__ float red[64];
    int i = blockIdx.x, h = blockIdx.y, b = blockIdx.z, t = threadIdx.x;

    qs[t] = Q[(((long)b*SEQ + i)*HH + h)*DHH + t];
    __syncthreads();

    for (int j = t; j < NJ; j += 64) {
        float s;
        if (j >= i + 2) {
            s = NEGF;
        } else {
            const float* kr = Kb + ((long)b*NJ + j)*DHH;
            float d = 0.f;
            #pragma unroll 8
            for (int e = 0; e < DHH; ++e) d += qs[e] * kr[e];
            int m = i - j; if (m < 0) m = 0;
            int bucket;
            if (m < 16) bucket = m;
            else {
                int big = 16 + (int)(logf((float)m / 16.f) * (16.f / logf(8.f)));
                bucket = big < 31 ? big : 31;
            }
            s = d + relpos[bucket * HH + h];
        }
        sc[j] = s;
    }
    __syncthreads();

    float lm = NEGF;
    for (int j = t; j < NJ; j += 64) lm = fmaxf(lm, sc[j]);
    float mx = reduceMax(lm, red, 64);

    float ls = 0.f;
    for (int j = t; j < NJ; j += 64) ls += expf(sc[j] - mx);
    float denom = reduceSum(ls, red, 64);
    float inv = 1.f / denom;

    float acc = 0.f;
    for (int j = 0; j < NJ; ++j) {
        float p = expf(sc[j] - mx) * inv;
        acc += p * Vb[((long)b*NJ + j)*DHH + t];
    }
    O[((long)b*SEQ + i)*DD + h*DHH + t] = acc;
}

// ---------------- GEGLU combine: g1 *= silu(g2) ----------------
__global__ __launch_bounds__(256)
void geglu_mul_kernel(float* __restrict__ g1, const float* __restrict__ g2, long count)
{
    long base = (long)blockIdx.x * 1024 + threadIdx.x;
    #pragma unroll
    for (int e = 0; e < 4; ++e) {
        long idx = base + e*256;
        if (idx < count) g1[idx] = g1[idx] * siluf(g2[idx]);
    }
}

// ---------------- host orchestration ----------------
extern "C" void kernel_launch(void* const* d_in, const int* in_sizes, int n_in,
                              void* d_out, int out_size, void* d_ws, size_t ws_size,
                              hipStream_t stream)
{
    (void)in_sizes; (void)n_in; (void)out_size; (void)ws_size;

    const float* text_embed  = (const float*)d_in[0];
    const float* text_enc    = (const float*)d_in[1];
    const float* image_embed = (const float*)d_in[2];
    const float* timesteps   = (const float*)d_in[3];
    const float* te_w1 = (const float*)d_in[4];
    const float* te_b1 = (const float*)d_in[5];
    const float* te_w2 = (const float*)d_in[6];
    const float* te_b2 = (const float*)d_in[7];
    const float* t_w1  = (const float*)d_in[8];
    const float* t_b1  = (const float*)d_in[9];
    const float* t_w2  = (const float*)d_in[10];
    const float* t_b2  = (const float*)d_in[11];
    const float* t_w3  = (const float*)d_in[12];
    const float* t_b3  = (const float*)d_in[13];
    const float* lq    = (const float*)d_in[14];
    const float* relpos= (const float*)d_in[15];
    const float* norm_g= (const float*)d_in[16];
    const float* q_w   = (const float*)d_in[17];
    const float* kv_w  = (const float*)d_in[18];
    const float* null_kv=(const float*)d_in[19];
    const float* out_w = (const float*)d_in[20];
    const float* out_norm_g = (const float*)d_in[21];
    const float* ff_norm_g  = (const float*)d_in[22];
    const float* ff_w1 = (const float*)d_in[23];
    const float* ff_w2 = (const float*)d_in[24];
    const float* final_g = (const float*)d_in[25];
    const float* proj_w  = (const float*)d_in[26];
    float* out = (float*)d_out;

    // workspace layout (floats)
    float* ws = (float*)d_ws;
    const long NX = (long)NTOK * DD;          // 8,519,680
    float* X    = ws;                         // tokens / residual stream
    float* XN   = X  + NX;                    // LN output
    float* QT   = XN + NX;                    // q / out-proj tmp / geglu gate
    float* OB   = QT + NX;                    // te_h / o / geglu value
    float* KV   = OB + NX;                    // (NTOK,128)
    float* Kb   = KV + (long)NTOK*128;        // (B,261,64)
    float* Vb   = Kb + (long)BB*NJ*DHH;
    float* SEb  = Vb + (long)BB*NJ*DHH;       // (32,1024)
    float* H1b  = SEb + (long)BB*DD;          // (32,2048)
    float* H2b  = H1b + (long)BB*2*DD;
    float* TEMB = H2b + (long)BB*2*DD;        // (32,1024)
    float* FIN  = TEMB + (long)BB*DD;         // (32,1024)

    auto gemm = [&](const float* A, const float* Bm, const float* bias, float* C,
                    int M, int N, int K, int lda, int ldb, int ldc,
                    float alpha, int act, int accum) {
        dim3 g((N + BN - 1) / BN, (M + BM - 1) / BM);
        gemm_wmma_bf16<<<g, 256, 0, stream>>>(A, Bm, bias, C, M, N, K,
                                              lda, ldb, ldc, alpha, act, accum);
    };

    // ---- text encoder MLP: te = relu(enc@W1+b1)@W2+b2 ----
    gemm(text_enc, te_w1, te_b1, OB, BB*LL, DD, TEE, TEE, DD, DD, 1.f, 1, 0);
    gemm(OB, te_w2, te_b2, QT, BB*LL, DD, DD, DD, DD, DD, 1.f, 0, 0);

    // ---- time embedding MLP ----
    se_kernel<<<BB, 256, 0, stream>>>(timesteps, SEb);
    gemm(SEb, t_w1, t_b1, H1b, BB, 2*DD, DD, DD, 2*DD, 2*DD, 1.f, 2, 0);
    gemm(H1b, t_w2, t_b2, H2b, BB, 2*DD, 2*DD, 2*DD, 2*DD, 2*DD, 1.f, 2, 0);
    gemm(H2b, t_w3, t_b3, TEMB, BB, DD, 2*DD, 2*DD, DD, DD, 1.f, 0, 0);

    // ---- assemble token stream X (B,260,D) ----
    assemble_kernel<<<dim3(SEQ, BB), 256, 0, stream>>>(QT, text_embed, TEMB, image_embed, lq, X);

    // ---- transformer layers ----
    for (int l = 0; l < 6; ++l) {
        ln_kernel<<<NTOK, 256, 0, stream>>>(X, norm_g + (long)l*DD, nullptr, XN, DD, 0, DD, 0);

        gemm(XN, q_w + (long)l*DD*DD, nullptr, QT, NTOK, DD, DD, DD, DD, DD, 16.f, 0, 0);
        gemm(XN, kv_w + (long)l*DD*128, nullptr, KV, NTOK, 128, DD, DD, 128, 128, 1.f, 0, 0);

        rotary_q_kernel<<<dim3(HH, SEQ, BB), 64, 0, stream>>>(QT);
        nullkv_kernel<<<BB, 64, 0, stream>>>(null_kv + (long)l*2*DHH, Kb, Vb);
        rotary_kv_kernel<<<dim3(SEQ, BB), 64, 0, stream>>>(KV, Kb, Vb);

        attn_kernel<<<dim3(SEQ, HH, BB), 64, 0, stream>>>(QT, Kb, Vb, relpos, OB);

        gemm(OB, out_w + (long)l*DD*DD, nullptr, QT, NTOK, DD, DD, DD, DD, DD, 1.f, 0, 0);
        ln_kernel<<<NTOK, 256, 0, stream>>>(QT, out_norm_g + (long)l*DD, X, X, DD, DD, DD, 0);

        // FF (GEGLU), chunked over rows so ws stays small
        ln_kernel<<<NTOK, 256, 0, stream>>>(X, ff_norm_g + (long)l*DD, nullptr, XN, DD, 0, DD, 0);
        const int CH = NTOK / 4;   // 2080
        for (int c = 0; c < 4; ++c) {
            const float* fnc = XN + (long)c*CH*DD;
            const float* w1  = ff_w1 + (long)l*DD*2*FFN;
            gemm(fnc, w1,        nullptr, OB, CH, FFN, DD, DD, 2*FFN, FFN, 1.f, 0, 0);
            gemm(fnc, w1 + FFN,  nullptr, QT, CH, FFN, DD, DD, 2*FFN, FFN, 1.f, 0, 0);
            long cnt = (long)CH * FFN;
            geglu_mul_kernel<<<(unsigned)((cnt + 1023) / 1024), 256, 0, stream>>>(OB, QT, cnt);
            gemm(OB, ff_w2 + (long)l*FFN*DD, nullptr, X + (long)c*CH*DD,
                 CH, DD, FFN, FFN, DD, DD, 1.f, 0, 1 /*accumulate residual*/);
        }
    }

    // ---- final: stable LN on last token, project ----
    ln_kernel<<<BB, 256, 0, stream>>>(X + (long)(SEQ - 1)*DD, final_g, nullptr, FIN,
                                      (long)SEQ*DD, 0, DD, 1 /*stable*/);
    gemm(FIN, proj_w, nullptr, out, BB, DD, DD, DD, DD, DD, 1.f, 0, 0);
}
